// EfficientSelfAttention_87582973100414
// MI455X (gfx1250) — matrix operand order repaired and checked by reading, not compiled, not run on hardware
//
#include <hip/hip_runtime.h>
#include <hip/hip_bf16.h>
#include <stdint.h>

// Problem constants (compile-time, from the reference)
#define BB 4
#define TT 8192
#define DD 1024
#define HH 16
#define HD 64
#define MM (BB * TT)          // 32768 rows

typedef __attribute__((ext_vector_type(8)))  __bf16 v8bf;
typedef __attribute__((ext_vector_type(16))) __bf16 v16bf;
typedef __attribute__((ext_vector_type(8)))  float  v8f;

static __device__ __forceinline__ float bf2f(unsigned short u) {
  unsigned int x = ((unsigned int)u) << 16;
  return __builtin_bit_cast(float, x);
}
static __device__ __forceinline__ unsigned short f2bf(float f) {
  unsigned int x = __builtin_bit_cast(unsigned int, f);
  x += 0x7FFFu + ((x >> 16) & 1u);      // round-to-nearest-even
  return (unsigned short)(x >> 16);
}

// --- WMMA fragment loaders (bf16, 16x16x32) ------------------------------
// A 16x32 (MxK): lane = hs*16 + (m&15); elems e -> K = hs*16 + e
// (16 contiguous, matching B).  p points at &A[m][k0].
static __device__ __forceinline__ v16bf load_a16(const unsigned short* p, int hs) {
  v8bf lo = *(const v8bf*)(p + hs * 16);
  v8bf hi = *(const v8bf*)(p + hs * 16 + 8);
  return __builtin_shufflevector(lo, hi, 0,1,2,3,4,5,6,7,8,9,10,11,12,13,14,15);
}
// B 32x16 (KxN): lane n = lane&15, elems e -> K = hs*16 + e (16 contiguous).
// p points at &Bt[n][k0] where Bt is stored N-major (row n contiguous in K).
static __device__ __forceinline__ v16bf load_b16(const unsigned short* p, int hs) {
  v8bf lo = *(const v8bf*)(p + hs * 16);
  v8bf hi = *(const v8bf*)(p + hs * 16 + 8);
  return __builtin_shufflevector(lo, hi, 0,1,2,3,4,5,6,7,8,9,10,11,12,13,14,15);
}
static __device__ __forceinline__ v8f wmma_bf16(v16bf a, v16bf b, v8f c) {
  return __builtin_amdgcn_wmma_f32_16x16x32_bf16(false, a, false, b, (short)0, c,
                                                 false, false);
}

// --- Kernel 1: fp32 -> bf16 convert ---------------------------------------
__global__ void cvt_f32_bf16(const float* __restrict__ src,
                             unsigned short* __restrict__ dst, int n) {
  int i = blockIdx.x * blockDim.x + threadIdx.x;
  if (i < n) dst[i] = f2bf(src[i]);
}

// --- Kernel 2: LayerNorm over D=1024 + bf16 cast of x and nx --------------
__global__ void ln_cast(const float* __restrict__ x,
                        const float* __restrict__ gamma,
                        const float* __restrict__ beta,
                        unsigned short* __restrict__ nxb,
                        unsigned short* __restrict__ xb) {
  __shared__ float r1[256];
  __shared__ float r2[256];
  const int row = blockIdx.x;           // 0..MM-1
  const int tid = threadIdx.x;          // 256 threads, 4 elems each
  const float4 xv = ((const float4*)(x + (size_t)row * DD))[tid];
  float s  = xv.x + xv.y + xv.z + xv.w;
  float s2 = xv.x * xv.x + xv.y * xv.y + xv.z * xv.z + xv.w * xv.w;
  r1[tid] = s; r2[tid] = s2;
  __syncthreads();
  for (int off = 128; off > 0; off >>= 1) {
    if (tid < off) { r1[tid] += r1[tid + off]; r2[tid] += r2[tid + off]; }
    __syncthreads();
  }
  const float mu  = r1[0] * (1.0f / DD);
  const float var = r2[0] * (1.0f / DD) - mu * mu;
  const float rs  = rsqrtf(var + 1e-5f);
  const int base = tid * 4;
  const float xs[4] = {xv.x, xv.y, xv.z, xv.w};
  #pragma unroll
  for (int c = 0; c < 4; ++c) {
    const int col = base + c;
    const float nv = (xs[c] - mu) * rs * gamma[col] + beta[col];
    nxb[(size_t)row * DD + col] = f2bf(nv);
    xb [(size_t)row * DD + col] = f2bf(xs[c]);
  }
}

// --- Kernel 3: GEMM  C[m,n] = sum_k A[m,k]*W[n,k] + bias[n] ---------------
// A: [M x K] bf16 row-major; W: [N x K] bf16 row-major (torch Linear weight,
// so this computes A @ W.T).
// Block = 256 threads (8 waves); wave tile = 32(m) x 64(n); block = 256 x 64.
// The shared 64x32 B-tile is double-buffered in LDS via async-to-LDS DMA
// (256 threads x b128 = 4KB tile, ASYNCcnt-tracked).  The last k-step is
// peeled so the steady-state loop issues the next DMA unconditionally.
// grid = (M/256, N/64).
__global__ void gemm_bias(const unsigned short* __restrict__ A,
                          const unsigned short* __restrict__ W,
                          const float* __restrict__ bias,
                          unsigned short* __restrict__ outb,   // bf16 or null
                          float* __restrict__ outf,            // fp32 or null
                          int K, int N) {
  __shared__ unsigned short bsh[2][64 * 32];   // 2 x 4KB B tiles
  const int tid  = threadIdx.x;
  const int wave = tid >> 5;
  const int lane = tid & 31;
  const int lr = lane & 15, hs = lane >> 4;
  const int mbase = blockIdx.x * 256 + wave * 32;
  const int nbase = blockIdx.y * 64;

  // Async staging: this thread owns one 16B chunk of the 64x32 B tile.
  const int brow = tid >> 2;                 // 0..63 (n within tile)
  const int bseg = tid & 3;                  // 0..3  (8 ushorts each)
  const unsigned short* gB = W + (size_t)(nbase + brow) * K + bseg * 8;
  const unsigned lds0 = (unsigned)(size_t)&bsh[0][brow * 32 + bseg * 8];
  const unsigned lds1 = (unsigned)(size_t)&bsh[1][brow * 32 + bseg * 8];

  v8f acc[2][4] = {};
  const unsigned short* ap0 = A + (size_t)(mbase + lr) * K;
  const unsigned short* ap1 = A + (size_t)(mbase + 16 + lr) * K;

  const int nsteps = K / 32;

  // prologue: stage k0=0 into buffer 0
  asm volatile("global_load_async_to_lds_b128 %0, %1, off"
               :: "v"(lds0), "v"(gB) : "memory");

  // steady state: always issue next tile, wait for current, compute
  for (int step = 0; step < nsteps - 1; ++step) {
    const int k0  = step * 32;
    const int cur = step & 1;
    {
      const unsigned dst = cur ? lds0 : lds1;
      const unsigned short* src = gB + (size_t)(step + 1) * 32;
      asm volatile("global_load_async_to_lds_b128 %0, %1, off"
                   :: "v"(dst), "v"(src) : "memory");
    }
    asm volatile("s_wait_asynccnt 0x1" ::: "memory");  // current tile landed
    __syncthreads();   // all waves' current-tile DMA complete

    __builtin_prefetch(ap0 + k0 + 256, 0, 1);   // global_prefetch_b8
    const v16bf a0 = load_a16(ap0 + k0, hs);
    const v16bf a1 = load_a16(ap1 + k0, hs);
    const unsigned short* bb = &bsh[cur][0];
    #pragma unroll
    for (int j = 0; j < 4; ++j) {
      const unsigned short* bp = bb + (j * 16 + lr) * 32;   // row n, 32 k elems
      v8bf lo = *(const v8bf*)(bp + hs * 16);               // ds_load_b128
      v8bf hi = *(const v8bf*)(bp + hs * 16 + 8);
      const v16bf bfr = __builtin_shufflevector(
          lo, hi, 0,1,2,3,4,5,6,7,8,9,10,11,12,13,14,15);
      acc[0][j] = wmma_bf16(a0, bfr, acc[0][j]);
      acc[1][j] = wmma_bf16(a1, bfr, acc[1][j]);
    }
    __syncthreads();   // finished reading cur before it is overwritten
  }

  // peeled final k-step: no further DMA, drain ASYNCcnt to zero
  {
    const int step = nsteps - 1;
    const int k0  = step * 32;
    const int cur = step & 1;
    asm volatile("s_wait_asynccnt 0x0" ::: "memory");
    __syncthreads();

    const v16bf a0 = load_a16(ap0 + k0, hs);
    const v16bf a1 = load_a16(ap1 + k0, hs);
    const unsigned short* bb = &bsh[cur][0];
    #pragma unroll
    for (int j = 0; j < 4; ++j) {
      const unsigned short* bp = bb + (j * 16 + lr) * 32;
      v8bf lo = *(const v8bf*)(bp + hs * 16);
      v8bf hi = *(const v8bf*)(bp + hs * 16 + 8);
      const v16bf bfr = __builtin_shufflevector(
          lo, hi, 0,1,2,3,4,5,6,7,8,9,10,11,12,13,14,15);
      acc[0][j] = wmma_bf16(a0, bfr, acc[0][j]);
      acc[1][j] = wmma_bf16(a1, bfr, acc[1][j]);
    }
  }

  #pragma unroll
  for (int mi = 0; mi < 2; ++mi) {
    #pragma unroll
    for (int j = 0; j < 4; ++j) {
      const int n = nbase + j * 16 + lr;
      const float bv = bias[n];
      #pragma unroll
      for (int r = 0; r < 8; ++r) {
        const int m = mbase + mi * 16 + r + hs * 8;
        const float v = acc[mi][j][r] + bv;
        const size_t idx = (size_t)m * N + n;
        if (outf) outf[idx] = v;
        else      outb[idx] = f2bf(v);
      }
    }
  }
}

// --- Kernel 4: per-head softmax (q,k) + pack/transpose (k,v) --------------
// One wave per (b,t,h).  q stays row-major [B*T, D]; k,v become [bh][64][T].
__global__ void softmax_pack(const unsigned short* __restrict__ qraw,
                             const unsigned short* __restrict__ kraw,
                             const unsigned short* __restrict__ vraw,
                             unsigned short* __restrict__ qb,
                             unsigned short* __restrict__ kT,
                             unsigned short* __restrict__ vT) {
  const int gw   = (blockIdx.x * blockDim.x + threadIdx.x) >> 5; // 0..B*T*H-1
  const int lane = threadIdx.x & 31;
  const int h  = gw % HH;
  const int bt = gw / HH;           // b*T + t
  const int b  = bt / TT;
  const int t  = bt % TT;
  const int bh = b * HH + h;
  const size_t rowoff = (size_t)bt * DD + h * HD;
  const size_t toff0 = ((size_t)bh * HD + lane)      * TT + t;
  const size_t toff1 = ((size_t)bh * HD + lane + 32) * TT + t;

  // q softmax over 64 (2 values per lane)
  {
    float v0 = bf2f(qraw[rowoff + lane]);
    float v1 = bf2f(qraw[rowoff + 32 + lane]);
    float m = fmaxf(v0, v1);
    for (int off = 16; off > 0; off >>= 1) m = fmaxf(m, __shfl_xor(m, off));
    float e0 = __expf(v0 - m), e1 = __expf(v1 - m);
    float s = e0 + e1;
    for (int off = 16; off > 0; off >>= 1) s += __shfl_xor(s, off);
    const float inv = 1.0f / s;
    qb[rowoff + lane]      = f2bf(e0 * inv);
    qb[rowoff + 32 + lane] = f2bf(e1 * inv);
  }
  // k softmax over 64, stored transposed [bh][d][t]
  {
    float v0 = bf2f(kraw[rowoff + lane]);
    float v1 = bf2f(kraw[rowoff + 32 + lane]);
    float m = fmaxf(v0, v1);
    for (int off = 16; off > 0; off >>= 1) m = fmaxf(m, __shfl_xor(m, off));
    float e0 = __expf(v0 - m), e1 = __expf(v1 - m);
    float s = e0 + e1;
    for (int off = 16; off > 0; off >>= 1) s += __shfl_xor(s, off);
    const float inv = 1.0f / s;
    kT[toff0] = f2bf(e0 * inv);
    kT[toff1] = f2bf(e1 * inv);
  }
  // v transpose only
  vT[toff0] = vraw[rowoff + lane];
  vT[toff1] = vraw[rowoff + 32 + lane];
}

// --- Kernel 5: context = K^T V per (b,h): [64 x 64] -----------------------
// 1 block per bh, 8 waves x 1024 tokens each, LDS fp32 reduction.
// Output stored transposed: ctxT[bh][l][d] (so it is B-layout for kernel 6).
__global__ void context_gemm(const unsigned short* __restrict__ kT,
                             const unsigned short* __restrict__ vT,
                             unsigned short* __restrict__ ctxT) {
  __shared__ float lds[HD * HD];
  const int bh   = blockIdx.x;
  const int tid  = threadIdx.x;
  const int wave = tid >> 5;
  const int lane = tid & 31;
  const int lr = lane & 15, hs = lane >> 4;

  for (int i = tid; i < HD * HD; i += 256) lds[i] = 0.0f;
  __syncthreads();

  const unsigned short* kb = kT + (size_t)bh * HD * TT;
  const unsigned short* vb = vT + (size_t)bh * HD * TT;
  v8f acc[4][4] = {};

  const int tbeg = wave * (TT / 8);
  for (int t0 = tbeg; t0 < tbeg + (TT / 8); t0 += 32) {
    v16bf bfrag[4];
    #pragma unroll
    for (int ni = 0; ni < 4; ++ni)
      bfrag[ni] = load_b16(vb + (size_t)(ni * 16 + lr) * TT + t0, hs);
    #pragma unroll
    for (int mi = 0; mi < 4; ++mi) {
      const v16bf a = load_a16(kb + (size_t)(mi * 16 + lr) * TT + t0, hs);
      #pragma unroll
      for (int ni = 0; ni < 4; ++ni)
        acc[mi][ni] = wmma_bf16(a, bfrag[ni], acc[mi][ni]);
    }
  }

  #pragma unroll
  for (int mi = 0; mi < 4; ++mi)
    #pragma unroll
    for (int ni = 0; ni < 4; ++ni)
      #pragma unroll
      for (int r = 0; r < 8; ++r) {
        const int d = mi * 16 + r + hs * 8;
        const int l = ni * 16 + lr;
        atomicAdd(&lds[d * HD + l], acc[mi][ni][r]);   // ds_add_f32
      }
  __syncthreads();

  for (int i = tid; i < HD * HD; i += 256) {
    const int l = i >> 6, d = i & 63;
    ctxT[(size_t)bh * HD * HD + i] = f2bf(lds[d * HD + l]);  // [l][d]
  }
}

// --- Kernel 6: y = Q @ ctx ; out = h + y ---------------------------------
// wave = one (b,h, 16-token) tile, N = 64.  waves total = B*H*T/16.
__global__ void out_gemm(const unsigned short* __restrict__ qb,
                         const unsigned short* __restrict__ ctxT,
                         const float* __restrict__ hraw,
                         float* __restrict__ out) {
  const int gw   = (blockIdx.x * blockDim.x + threadIdx.x) >> 5;
  const int lane = threadIdx.x & 31;
  const int lr = lane & 15, hs = lane >> 4;
  const int ntt = TT / 16;
  const int tt = gw % ntt;
  const int bh = gw / ntt;
  const int h = bh % HH, b = bh / HH;
  const int tb = tt * 16;

  const unsigned short* ap = qb + ((size_t)(b * TT + tb + lr)) * DD + h * HD;
  const unsigned short* cb = ctxT + (size_t)bh * HD * HD;

  v8f acc[4] = {};
  #pragma unroll
  for (int k0 = 0; k0 < HD; k0 += 32) {
    const v16bf a = load_a16(ap + k0, hs);
    #pragma unroll
    for (int j = 0; j < 4; ++j) {
      const v16bf bf = load_b16(cb + (size_t)(j * 16 + lr) * HD + k0, hs);
      acc[j] = wmma_bf16(a, bf, acc[j]);
    }
  }

  #pragma unroll
  for (int j = 0; j < 4; ++j) {
    const int n = h * HD + j * 16 + lr;
    #pragma unroll
    for (int r = 0; r < 8; ++r) {
      const int m = tb + r + hs * 8;
      const size_t idx = (size_t)(b * TT + m) * DD + n;
      out[idx] = hraw[idx] + acc[j][r];
    }
  }
}

// --------------------------------------------------------------------------
extern "C" void kernel_launch(void* const* d_in, const int* in_sizes, int n_in,
                              void* d_out, int out_size, void* d_ws, size_t ws_size,
                              hipStream_t stream) {
  const float* x   = (const float*)d_in[0];
  const float* Wq  = (const float*)d_in[1];
  const float* bq  = (const float*)d_in[2];
  const float* Wk  = (const float*)d_in[3];
  const float* bk  = (const float*)d_in[4];
  const float* Wv  = (const float*)d_in[5];
  const float* bv  = (const float*)d_in[6];
  const float* Wh  = (const float*)d_in[7];
  const float* bhp = (const float*)d_in[8];
  const float* gamma = (const float*)d_in[9];
  const float* beta  = (const float*)d_in[10];
  float* out = (float*)d_out;

  // Workspace carve-up (256B aligned slices)
  char* w = (char*)d_ws;
  auto alloc = [&](size_t bytes) -> void* {
    void* p = (void*)w;
    w += (bytes + 255) & ~(size_t)255;
    return p;
  };
  const size_t WSZ = (size_t)DD * DD;       // 1M weights
  const size_t ASZ = (size_t)MM * DD;       // 32M activations
  unsigned short* Wqb = (unsigned short*)alloc(WSZ * 2);
  unsigned short* Wkb = (unsigned short*)alloc(WSZ * 2);
  unsigned short* Wvb = (unsigned short*)alloc(WSZ * 2);
  unsigned short* Whb = (unsigned short*)alloc(WSZ * 2);
  unsigned short* nxb = (unsigned short*)alloc(ASZ * 2);
  unsigned short* xb  = (unsigned short*)alloc(ASZ * 2);
  unsigned short* qraw = (unsigned short*)alloc(ASZ * 2);
  unsigned short* kraw = (unsigned short*)alloc(ASZ * 2);
  unsigned short* vraw = (unsigned short*)alloc(ASZ * 2);
  float*          hraw = (float*)alloc(ASZ * 4);
  unsigned short* qbuf = (unsigned short*)alloc(ASZ * 2);
  unsigned short* kTb  = (unsigned short*)alloc(ASZ * 2);
  unsigned short* vTb  = (unsigned short*)alloc(ASZ * 2);
  unsigned short* ctxT = (unsigned short*)alloc((size_t)BB * HH * HD * HD * 2);

  // 1) weights -> bf16
  cvt_f32_bf16<<<(int)(WSZ / 256), 256, 0, stream>>>(Wq, Wqb, (int)WSZ);
  cvt_f32_bf16<<<(int)(WSZ / 256), 256, 0, stream>>>(Wk, Wkb, (int)WSZ);
  cvt_f32_bf16<<<(int)(WSZ / 256), 256, 0, stream>>>(Wv, Wvb, (int)WSZ);
  cvt_f32_bf16<<<(int)(WSZ / 256), 256, 0, stream>>>(Wh, Whb, (int)WSZ);

  // 2) LayerNorm + casts
  ln_cast<<<MM, 256, 0, stream>>>(x, gamma, beta, nxb, xb);

  // 3) Four big WMMA GEMMs (M=32768, N=K=1024), async-LDS staged weights
  dim3 ggrid(MM / 256, DD / 64);
  gemm_bias<<<ggrid, 256, 0, stream>>>(nxb, Wqb, bq,  qraw, nullptr, DD, DD);
  gemm_bias<<<ggrid, 256, 0, stream>>>(nxb, Wkb, bk,  kraw, nullptr, DD, DD);
  gemm_bias<<<ggrid, 256, 0, stream>>>(nxb, Wvb, bv,  vraw, nullptr, DD, DD);
  gemm_bias<<<ggrid, 256, 0, stream>>>(xb,  Whb, bhp, nullptr, hraw, DD, DD);

  // 4) softmax over head_dim + transposed packing of k,v
  const int smx_waves = BB * TT * HH;                 // 524288
  softmax_pack<<<smx_waves / 8, 256, 0, stream>>>(qraw, kraw, vraw,
                                                  qbuf, kTb, vTb);

  // 5) context = K^T V per (b,h)
  context_gemm<<<BB * HH, 256, 0, stream>>>(kTb, vTb, ctxT);

  // 6) out = x@Wh.T + b + Q@ctx
  const int out_waves = BB * HH * (TT / 16);          // 32768
  out_gemm<<<out_waves / 8, 256, 0, stream>>>(qbuf, ctxT, hraw, out);
}
